// SimpleCnnNcpActor_28381143892325
// MI455X (gfx1250) — matrix operand order repaired and checked by reading, not compiled, not run on hardware
//
#include <hip/hip_runtime.h>
#include <hip/hip_bf16.h>
#include <hip/hip_fp16.h>
#include <stdint.h>

// ---------------------------------------------------------------------------
// SimpleCnnNcpActor for MI455X (gfx1250, wave32, WMMA).
// conv1 FUSED into conv2; conv2/3/4 are implicit-GEMM v_wmma_f32_16x16x32_f16
// with dual accumulators. The FULL im2col B operand for a block (<=36KB) is
// staged into LDS in fragment order ONCE, so the WMMA K loop has no address
// arithmetic, no barriers: just v16h A loads + ds_load_b128 B loads + wmma.
// LTC recurrence + head in one kernel per batch, async global->LDS staging.
// ---------------------------------------------------------------------------

typedef __attribute__((ext_vector_type(16))) _Float16 v16h;
typedef __attribute__((ext_vector_type(8)))  float    v8f;

#define FRAMES 256   // B*S = 32*8

#define AS1 __attribute__((address_space(1)))
#define AS3 __attribute__((address_space(3)))

#if defined(__has_builtin)
#if __has_builtin(__builtin_amdgcn_global_load_async_to_lds_b32) && \
    __has_builtin(__builtin_amdgcn_s_wait_asynccnt)
#define USE_ASYNC_LDS 1
#endif
#endif
#ifndef USE_ASYNC_LDS
#define USE_ASYNC_LDS 0
#endif

// ---------------------------------------------------------------------------
// Weight pre-swizzle: f32 conv weights [OC][IC][3][3] -> f16 fragments in the
// 16-bit A-matrix (16x32) operand order used here (contiguous k, same as B):
//   lane L holds row M = mt*16 + (L&15)
//   half e: kLocal = (L>=16 ? 16:0) + e
// Stored frag[((mt*KSTEPS+kt)*32 + lane)*16 + e] => one aligned v16h per wave.
// ---------------------------------------------------------------------------
__global__ void wfrag_prep(const float* __restrict__ w, _Float16* __restrict__ frag,
                           int OC, int IC, int mtiles, int ksteps) {
  const int K = IC * 9;
  const int total = mtiles * ksteps * 512;
  for (int t = blockIdx.x * blockDim.x + threadIdx.x; t < total;
       t += gridDim.x * blockDim.x) {
    int e    = t & 15;
    int lane = (t >> 4) & 31;
    int kt   = (t >> 9) % ksteps;
    int mt   = t / (512 * ksteps);
    int m    = mt * 16 + (lane & 15);
    int v    = e >> 1, j = e & 1;
    int klocal = ((lane & 16) ? 16 : 0) + 2 * v + j;
    int k = kt * 32 + klocal;
    float val = 0.f;
    if (k < K && m < OC) {
      int ic = k / 9, r = k - ic * 9;
      val = w[(m * IC + ic) * 9 + r];
    }
    frag[t] = (_Float16)val;
  }
}

// ---------------------------------------------------------------------------
// Fused conv1+conv2. One block = (frame, 16-pixel conv2 tile), 128 threads.
//  Stage A: depth window (11x82, zero-padded) + w1/b1 -> LDS
//  Stage B: conv1 computed DIRECTLY into the fragment-ordered im2col buffer
//           (36 ksteps x 512 halfs = 36KB). All k->(ic,ky,kx) math lives here.
//  Stage C: barrier-free WMMA K loop, dual accumulators; B operand is two
//           ds_load_b128 per step at a linearly advancing LDS address.
// conv2 geom: 128ic, 64x40 in, 64oc, 32x20 out, K=1152.
// ---------------------------------------------------------------------------
__global__ __launch_bounds__(128) void conv12_fused(
    const float* __restrict__ depth, const float* __restrict__ w1,
    const float* __restrict__ b1, const _Float16* __restrict__ wfrag,
    const float* __restrict__ b2, _Float16* __restrict__ out) {
  constexpr int KSTEPS = 36;          // K = 128*9 = 1152 (even # of steps)
  constexpr int OW = 20;              // conv2 out 32x20, NPIX=640, NTILES=40
  __shared__ float depth_s[11][82];
  __shared__ float w1_s[128 * 9];
  __shared__ float b1_s[128];
  __shared__ alignas(32) _Float16 lds_B[KSTEPS * 512];   // 36 KB im2col

  const int f    = blockIdx.x / 40;
  const int nt   = blockIdx.x % 40;
  const int wave = threadIdx.x >> 5;
  const int lane = threadIdx.x & 31;
  const int tid  = threadIdx.x;

  const int p0   = nt * 16;
  const int oy0  = p0 / OW;
  const int iy0r = 2 * oy0 - 1;           // first conv1 row covered by tile
  const int by2  = 2 * iy0r - 1;          // first depth row in LDS window

  // Stage A: depth window + conv1 weights -> LDS
  for (int t = tid; t < 11 * 82; t += 128) {
    int lr = t / 82, lc = t - lr * 82;
    int gy = by2 + lr, gx = lc - 1;
    float v = 0.f;
    if (gy >= 0 && gy < 128 && gx >= 0 && gx < 80)
      v = depth[((size_t)f * 128 + gy) * 80 + gx];
    depth_s[lr][lc] = v;
  }
  for (int t = tid; t < 128 * 9; t += 128) w1_s[t] = w1[t];
  b1_s[tid] = b1[tid];
  __syncthreads();

  // Stage B: conv1 -> im2col fragments (one-time address math + 9-tap MACs)
  for (int t = tid; t < KSTEPS * 512; t += 128) {
    int sl = t & 511;
    int kt = t >> 9;
    int ls = sl >> 4, el = sl & 15;
    int n  = ls & 15;
    int k  = kt * 32 + ((ls & 16) ? 16 : 0) + el;       // < 1152 always
    int ic = k / 9, r = k - ic * 9;
    int ky = r / 3, kx = r - ky * 3;
    int p  = p0 + n;
    int oy = p / OW, ox = p - oy * OW;
    int iy = 2 * oy - 1 + ky, ix = 2 * ox - 1 + kx;     // conv1 output coords
    _Float16 val = (_Float16)0.f;
    if (iy >= 0 && iy < 64 && ix >= 0 && ix < 40) {
      float acc = b1_s[ic];
      int dr = 2 * (iy - iy0r);                          // depth_s row of ty=0
      int dc = 2 * ix;                                   // depth_s col of tx=0
#pragma unroll
      for (int ty = 0; ty < 3; ++ty)
#pragma unroll
        for (int tx = 0; tx < 3; ++tx)
          acc += w1_s[ic * 9 + ty * 3 + tx] * depth_s[dr + ty][dc + tx];
      val = (_Float16)(acc > 0.f ? acc : 0.f);
    }
    lds_B[t] = val;
  }
  __syncthreads();

  // Stage C: pure WMMA loop (no barriers, no address math)
  const _Float16* bptr = lds_B + lane * 16;
  auto loadA = [&](int kt) {
    return *(const v16h*)(wfrag + ((size_t)(wave * KSTEPS + kt) * 32 + lane) * 16);
  };

  v8f acc0 = {}, acc1 = {};
  for (int kt = 0; kt < KSTEPS; kt += 2) {
    if (kt + 2 < KSTEPS)  // global_prefetch_b8 on upcoming A fragments
      __builtin_prefetch((const void*)(wfrag +
          ((size_t)(wave * KSTEPS + kt + 2) * 32 + lane) * 16), 0, 1);

    v16h a0 = loadA(kt);
    v16h b0 = *(const v16h*)(bptr + (size_t)kt * 512);
    v16h a1 = loadA(kt + 1);
    v16h b1v = *(const v16h*)(bptr + (size_t)(kt + 1) * 512);
    acc0 = __builtin_amdgcn_wmma_f32_16x16x32_f16(false, a0, false, b0,
                                                  (short)0, acc0, false, false);
    acc1 = __builtin_amdgcn_wmma_f32_16x16x32_f16(false, a1, false, b1v,
                                                  (short)0, acc1, false, false);
  }

  // D layout: VGPR r -> M = wave*16 + r + 8*(lane>=16), N = lane&15.
  const int p = p0 + (lane & 15);
#pragma unroll
  for (int r = 0; r < 8; ++r) {
    int m = wave * 16 + r + ((lane & 16) ? 8 : 0);
    float vv = acc0[r] + acc1[r] + b2[m];
    vv = vv > 0.f ? vv : 0.f;
    out[((size_t)f * 64 + m) * 640 + p] = (_Float16)vv;
  }
}

// ---------------------------------------------------------------------------
// Implicit-GEMM conv (global f16 input) for conv3/conv4. The FULL im2col B
// operand (KSTEPS*512 halfs = 18KB) is staged to LDS once (single barrier),
// then a barrier-free dual-accumulator WMMA loop.
// ---------------------------------------------------------------------------
template <typename OutT, int IC, int IH, int IW, int OC, int OH, int OW>
__global__ void conv_wmma(const _Float16* __restrict__ in,
                          const _Float16* __restrict__ wfrag,
                          const float* __restrict__ bias,
                          OutT* __restrict__ out) {
  constexpr int K      = IC * 9;
  constexpr int KSTEPS = (K + 31) / 32;   // 18 (even) for both instantiations
  constexpr int NPIX   = OH * OW;
  constexpr int NTILES = (NPIX + 15) / 16;

  __shared__ alignas(32) _Float16 lds_B[KSTEPS * 512];   // 18 KB im2col

  const int f    = blockIdx.x / NTILES;
  const int nt   = blockIdx.x % NTILES;
  const int wave = threadIdx.x >> 5;
  const int lane = threadIdx.x & 31;
  const _Float16* inF = in + (size_t)f * IC * IH * IW;

  // one-shot im2col staging (all K-steps)
  for (int t = threadIdx.x; t < KSTEPS * 512; t += blockDim.x) {
    int sl = t & 511;
    int kt = t >> 9;
    int ls = sl >> 4, el = sl & 15;
    int n  = ls & 15;
    int k  = kt * 32 + ((ls & 16) ? 16 : 0) + el;
    float val = 0.f;
    int p = nt * 16 + n;
    if (k < K && p < NPIX) {
      int ic = k / 9, r = k - ic * 9;
      int ky = r / 3, kx = r - ky * 3;
      int oy = p / OW, ox = p - oy * OW;
      int iy = 2 * oy - 1 + ky, ix = 2 * ox - 1 + kx;
      if (iy >= 0 && iy < IH && ix >= 0 && ix < IW)
        val = (float)inF[((size_t)ic * IH + iy) * IW + ix];
    }
    lds_B[t] = (_Float16)val;
  }
  __syncthreads();

  const _Float16* bptr = lds_B + lane * 16;
  auto loadA = [&](int kt) {
    return *(const v16h*)(wfrag + ((size_t)(wave * KSTEPS + kt) * 32 + lane) * 16);
  };

  v8f acc0 = {}, acc1 = {};
  for (int kt = 0; kt < KSTEPS; kt += 2) {
    if (kt + 2 < KSTEPS)
      __builtin_prefetch((const void*)(wfrag +
          ((size_t)(wave * KSTEPS + kt + 2) * 32 + lane) * 16), 0, 1);

    v16h a0 = loadA(kt);
    v16h b0 = *(const v16h*)(bptr + (size_t)kt * 512);
    v16h a1 = loadA(kt + 1);
    v16h b1v = *(const v16h*)(bptr + (size_t)(kt + 1) * 512);
    acc0 = __builtin_amdgcn_wmma_f32_16x16x32_f16(false, a0, false, b0,
                                                  (short)0, acc0, false, false);
    acc1 = __builtin_amdgcn_wmma_f32_16x16x32_f16(false, a1, false, b1v,
                                                  (short)0, acc1, false, false);
  }

  const int n = lane & 15;
  const int p = nt * 16 + n;
  if (p < NPIX) {
#pragma unroll
    for (int r = 0; r < 8; ++r) {
      int m = wave * 16 + r + ((lane & 16) ? 8 : 0);
      float vv = acc0[r] + acc1[r] + bias[m];
      vv = vv > 0.f ? vv : 0.f;
      out[((size_t)f * OC + m) * NPIX + p] = (OutT)vv;
    }
  }
}

// ---------------------------------------------------------------------------
// feats: conv4 out [f][32][8][5] flattened (c*40+y*5+x) ++ rel_pos[3],
// then inputs = feats*input_w + input_b.
// ---------------------------------------------------------------------------
__global__ void feats_build(const float* __restrict__ conv4out,
                            const float* __restrict__ relpos,
                            const float* __restrict__ iw, const float* __restrict__ ib,
                            float* __restrict__ inputs) {
  const int SE = 1283;
  const int total = FRAMES * SE;
  for (int t = blockIdx.x * blockDim.x + threadIdx.x; t < total;
       t += gridDim.x * blockDim.x) {
    int f = t / SE, j = t - f * SE;
    float v = (j < 1280) ? conv4out[(size_t)f * 1280 + j] : relpos[f * 3 + (j - 1280)];
    inputs[t] = v * iw[j] + ib[j];
  }
}

// ---------------------------------------------------------------------------
// NCP LTC recurrence + head. One block per batch element; blockDim (48,5).
// ---------------------------------------------------------------------------
__global__ __launch_bounds__(240) void ltc_kernel(
    const float* __restrict__ inputs,
    const float* __restrict__ gleak, const float* __restrict__ vleak,
    const float* __restrict__ cm,
    const float* __restrict__ mu, const float* __restrict__ sigma,
    const float* __restrict__ w_syn, const float* __restrict__ erev,
    const float* __restrict__ mask,
    const float* __restrict__ smu, const float* __restrict__ ssigma,
    const float* __restrict__ sw, const float* __restrict__ serev,
    const float* __restrict__ smask,
    const float* __restrict__ ow, const float* __restrict__ ob,
    const float* __restrict__ hw, const float* __restrict__ hb,
    float* __restrict__ d_out) {
  const int S = 8, U = 48, SE = 1283;
  const int b = blockIdx.x;
  const int u = threadIdx.x;                 // 0..47
  const int y = threadIdx.y;                 // 0..4
  const int tid = y * 48 + u;

  __shared__ float inp_s[1283];
  __shared__ float vsh[48];
  __shared__ float pn[5][48], pd[5][48];
  __shared__ float wnum_s[48], wden_s[48];
  __shared__ float opre[4];

  if (tid < U) vsh[tid] = 0.f;
  float acc_out = 0.f;

  for (int s = 0; s < S; ++s) {
    const float* inp = inputs + ((size_t)b * S + s) * SE;
#if USE_ASYNC_LDS
    // async global->LDS copy of this timestep's input vector (ASYNCcnt path)
    for (int i = tid; i < SE; i += 240)
      __builtin_amdgcn_global_load_async_to_lds_b32(
          (AS1 int*)(uintptr_t)(inp + i),
          (AS3 int*)(uint32_t)(uintptr_t)&inp_s[i], 0, 0);
    __builtin_amdgcn_s_wait_asynccnt(0);
#else
    for (int i = tid; i < SE; i += 240) inp_s[i] = inp[i];
#endif
    __syncthreads();

    // sensory synapses: reduce over 1283 inputs
    float an = 0.f, ad = 0.f;
    for (int i = y; i < SE; i += 5) {
      int idx = i * U + u;
      float x = sw[idx] / (1.f + __expf(-ssigma[idx] * (inp_s[i] - smu[idx]))) * smask[idx];
      an += x * serev[idx];
      ad += x;
    }
    pn[y][u] = an; pd[y][u] = ad;
    __syncthreads();
    if (y == 0) {
      float rn = 0.f, rd = 0.f;
#pragma unroll
      for (int yy = 0; yy < 5; ++yy) { rn += pn[yy][u]; rd += pd[yy][u]; }
      wnum_s[u] = rn; wden_s[u] = rd;
    }
    __syncthreads();

    // 6 ODE unfolds (sequential)
    for (int it = 0; it < 6; ++it) {
      float bn = 0.f, bd = 0.f;
      for (int j = y; j < U; j += 5) {
        int idx = j * U + u;
        float x = w_syn[idx] / (1.f + __expf(-sigma[idx] * (vsh[j] - mu[idx]))) * mask[idx];
        bn += x * erev[idx];
        bd += x;
      }
      pn[y][u] = bn; pd[y][u] = bd;
      __syncthreads();
      if (y == 0) {
        float rn = 0.f, rd = 0.f;
#pragma unroll
        for (int yy = 0; yy < 5; ++yy) { rn += pn[yy][u]; rd += pd[yy][u]; }
        float cmt = cm[u] * 6.f;  // ODE_UNFOLDS
        float num = cmt * vsh[u] + gleak[u] * vleak[u] + rn + wnum_s[u];
        float den = cmt + gleak[u] + rd + wden_s[u];
        pn[0][u] = num / (den + 1e-8f);
      }
      __syncthreads();
      if (y == 0) vsh[u] = pn[0][u];
      __syncthreads();
    }
    if (y == 0 && u < 4) acc_out += vsh[u] * ow[u] + ob[u];
    __syncthreads();
  }

  if (y == 0 && u < 4) opre[u] = acc_out * (1.f / 8.f);
  __syncthreads();
  if (tid < 2) {
    float sacc = hb[tid];
#pragma unroll
    for (int m = 0; m < 4; ++m) sacc += opre[m] * hw[m * 2 + tid];
    d_out[b * 2 + tid] = tanhf(sacc);
  }
}

// ---------------------------------------------------------------------------
extern "C" void kernel_launch(void* const* d_in, const int* in_sizes, int n_in,
                              void* d_out, int out_size, void* d_ws, size_t ws_size,
                              hipStream_t stream) {
  (void)in_sizes; (void)n_in; (void)out_size; (void)ws_size;
  const float* depth  = (const float*)d_in[0];
  const float* relpos = (const float*)d_in[1];
  const float* w1 = (const float*)d_in[2];  const float* b1 = (const float*)d_in[3];
  const float* w2 = (const float*)d_in[4];  const float* b2 = (const float*)d_in[5];
  const float* w3 = (const float*)d_in[6];  const float* b3 = (const float*)d_in[7];
  const float* w4 = (const float*)d_in[8];  const float* b4 = (const float*)d_in[9];
  const float* gleak = (const float*)d_in[10];
  const float* vleak = (const float*)d_in[11];
  const float* cm    = (const float*)d_in[12];
  const float* mu    = (const float*)d_in[13];
  const float* sigma = (const float*)d_in[14];
  const float* wsyn  = (const float*)d_in[15];
  const float* erev  = (const float*)d_in[16];
  const float* mask  = (const float*)d_in[17];
  const float* smu   = (const float*)d_in[18];
  const float* ssig  = (const float*)d_in[19];
  const float* sw    = (const float*)d_in[20];
  const float* serev = (const float*)d_in[21];
  const float* smask = (const float*)d_in[22];
  const float* iw    = (const float*)d_in[23];
  const float* ib    = (const float*)d_in[24];
  const float* ow    = (const float*)d_in[25];
  const float* ob    = (const float*)d_in[26];
  const float* hw    = (const float*)d_in[27];
  const float* hb    = (const float*)d_in[28];

  // workspace carve-up (256B aligned) — conv1 intermediate eliminated by fusion
  char* ws = (char*)d_ws;
  size_t off = 0;
  auto carve = [&](size_t bytes) {
    void* p = ws + off;
    off = (off + bytes + 255) & ~(size_t)255;
    return p;
  };
  _Float16* c2  = (_Float16*)carve((size_t)FRAMES * 64 * 32 * 20 * 2);   // 21 MB
  _Float16* c3  = (_Float16*)carve((size_t)FRAMES * 64 * 16 * 10 * 2);   // 5.2 MB
  float*    c4  = (float*)   carve((size_t)FRAMES * 32 * 8 * 5 * 4);     // 1.3 MB
  _Float16* wf2 = (_Float16*)carve((size_t)4 * 36 * 512 * 2);
  _Float16* wf3 = (_Float16*)carve((size_t)4 * 18 * 512 * 2);
  _Float16* wf4 = (_Float16*)carve((size_t)2 * 18 * 512 * 2);
  float*    inputs = (float*)carve((size_t)FRAMES * 1283 * 4);

  // weight fragment pre-swizzle (tiny)
  wfrag_prep<<<(4 * 36 * 512 + 255) / 256, 256, 0, stream>>>(w2, wf2, 64, 128, 4, 36);
  wfrag_prep<<<(4 * 18 * 512 + 255) / 256, 256, 0, stream>>>(w3, wf3, 64, 64, 4, 18);
  wfrag_prep<<<(2 * 18 * 512 + 255) / 256, 256, 0, stream>>>(w4, wf4, 32, 64, 2, 18);

  // fused conv1+conv2 (WMMA), then conv3/conv4 (WMMA)
  conv12_fused<<<FRAMES * 40, 128, 0, stream>>>(depth, w1, b1, wf2, b2, c2);
  conv_wmma<_Float16, 64, 32, 20, 64, 16, 10><<<FRAMES * 10, 128, 0, stream>>>(c2, wf3, b3, c3);
  conv_wmma<float,    64, 16, 10, 32,  8,  5><<<FRAMES *  3,  64, 0, stream>>>(c3, wf4, b4, c4);

  feats_build<<<(FRAMES * 1283 + 255) / 256, 256, 0, stream>>>(c4, relpos, iw, ib, inputs);

  ltc_kernel<<<32, dim3(48, 5), 0, stream>>>(inputs, gleak, vleak, cm, mu, sigma,
                                             wsyn, erev, mask, smu, ssig, sw, serev,
                                             smask, ow, ob, hw, hb, (float*)d_out);
}
